// Rconv_3D_5205500363257
// MI455X (gfx1250) — matrix operand order repaired and checked by reading, not compiled, not run on hardware
//
#include <hip/hip_runtime.h>
#include <hip/hip_bf16.h>
#include <cstdint>

typedef __attribute__((ext_vector_type(16))) _Float16 v16h;
typedef __attribute__((ext_vector_type(8)))  float    v8f;
typedef __attribute__((ext_vector_type(4)))  int      v4i;

typedef v4i __attribute__((address_space(1)))* as1_v4i_p;
typedef v4i __attribute__((address_space(3)))* as3_v4i_p;

#define LTOT 13824          // 24*24*24
#define CDIM 192
#define ODIM 576            // 3*192

#if __has_builtin(__builtin_amdgcn_global_load_async_to_lds_b128) && \
    __has_builtin(__builtin_amdgcn_s_wait_asynccnt)
#define USE_ASYNC_LDS 1
#else
#define USE_ASYNC_LDS 0
#endif

// ---------------------------------------------------------------------------
// WMMA GEMM: Out[o][l] = sum_c W[o][c] * X[c][l] + bias[o],  K = CDIM = 192.
// Block = 128 threads (4 waves). Block tile = 64(M) x 64(N).
// Wave w computes rows [my*64 + w*16, +16) x all 64 N columns:
//   4 f32 accumulators, A fragment shared across the 4 N sub-tiles.
// X slab [32K x 64N] staged in LDS via async global->LDS b128 copies.
// ---------------------------------------------------------------------------
__global__ __launch_bounds__(128) void wmma_gemm_kernel(
    const float* __restrict__ X,    // [CDIM][LTOT]
    const float* __restrict__ W,    // [O][CDIM]
    const float* __restrict__ bias, // [O]
    float* __restrict__ Out)        // [O][LTOT]
{
    __shared__ float tileX[32 * 64];           // 8 KB K-slab of X

    const int lane  = threadIdx.x & 31;
    const int wave  = threadIdx.x >> 5;
    const int l0    = blockIdx.x * 64;         // N tile base
    const int nlane = lane & 15;
    const int hi    = (lane >> 4) & 1;         // lane half

    // staging coords: 128 threads move 8 rows x 64 floats (=2KB) per pass
    const int srow = threadIdx.x >> 4;         // 0..7
    const int scol = (threadIdx.x & 15) * 4;   // 0,4,...,60

    v8f acc0 = {}, acc1 = {}, acc2 = {}, acc3 = {};

    // A k order used here (contiguous, same as B): lane<16 -> K 0..15 ; lane>=16 -> K 16..31
    const int arow  = blockIdx.y * 64 + wave * 16 + nlane;
    const int base1 = hi * 16;
    const int base2 = base1 + 8;

    #pragma unroll
    for (int kk = 0; kk < CDIM; kk += 32) {
        // ---- stage X[kk..kk+32)[l0..l0+64) into LDS --------------------
        #pragma unroll
        for (int p = 0; p < 4; ++p) {
            const int r = p * 8 + srow;
            const float* gp = X + (size_t)(kk + r) * LTOT + l0 + scol;
            float*       lp = &tileX[r * 64 + scol];
#if USE_ASYNC_LDS
            __builtin_amdgcn_global_load_async_to_lds_b128(
                (as1_v4i_p)(uintptr_t)gp, (as3_v4i_p)(uintptr_t)lp, 0, 0);
#else
            *(float4*)lp = *(const float4*)gp;
#endif
        }
#if USE_ASYNC_LDS
        __builtin_amdgcn_s_wait_asynccnt(0);
#endif
        __syncthreads();

        // ---- A fragment from global (weights, L2 resident) -------------
        const float* ap = W + (size_t)arow * CDIM + kk;
        float4 a0 = *(const float4*)(ap + base1);
        float4 a1 = *(const float4*)(ap + base1 + 4);
        float4 a2 = *(const float4*)(ap + base2);
        float4 a3 = *(const float4*)(ap + base2 + 4);
        v16h a;
        a[0]=(_Float16)a0.x; a[1]=(_Float16)a0.y; a[2]=(_Float16)a0.z; a[3]=(_Float16)a0.w;
        a[4]=(_Float16)a1.x; a[5]=(_Float16)a1.y; a[6]=(_Float16)a1.z; a[7]=(_Float16)a1.w;
        a[8]=(_Float16)a2.x; a[9]=(_Float16)a2.y; a[10]=(_Float16)a2.z; a[11]=(_Float16)a2.w;
        a[12]=(_Float16)a3.x; a[13]=(_Float16)a3.y; a[14]=(_Float16)a3.z; a[15]=(_Float16)a3.w;

        // ---- 4 B fragments from LDS, 4 WMMAs ---------------------------
        // B k order used here (contiguous, matches A): lane<16 holds K 0..15 of col n; lane>=16 holds K 16..31
        #pragma unroll
        for (int j = 0; j < 4; ++j) {
            const float* bp = &tileX[hi * 16 * 64 + j * 16 + nlane];
            v16h b;
            #pragma unroll
            for (int e = 0; e < 16; ++e) b[e] = (_Float16)bp[e * 64];
            v8f& acc = (j == 0) ? acc0 : (j == 1) ? acc1 : (j == 2) ? acc2 : acc3;
            acc = __builtin_amdgcn_wmma_f32_16x16x32_f16(
                false, a, false, b, (short)0, acc, false, false);
        }
        __syncthreads();   // protect LDS slab before next overwrite
    }

    // C/D layout: VGPR r -> (M=r, N=lane) lanes 0-15; (M=r+8, N=lane-16) lanes 16-31.
    const int rowb = blockIdx.y * 64 + wave * 16 + hi * 8;
    #pragma unroll
    for (int j = 0; j < 4; ++j) {
        const v8f& acc = (j == 0) ? acc0 : (j == 1) ? acc1 : (j == 2) ? acc2 : acc3;
        const int col = l0 + j * 16 + nlane;
        #pragma unroll
        for (int r = 0; r < 8; ++r)
            Out[(size_t)(rowb + r) * LTOT + col] = acc[r] + bias[rowb + r];
    }
}

// ---------------------------------------------------------------------------
// Depthwise 3x3x3 conv, zero padding, 576 channels. One thread per output.
// ---------------------------------------------------------------------------
__global__ __launch_bounds__(256) void dwconv_kernel(
    const float* __restrict__ in,   // [576][LTOT]
    const float* __restrict__ wdw,  // [576][27]
    const float* __restrict__ bdw,  // [576]
    float* __restrict__ out)        // [576][LTOT]
{
    int idx = blockIdx.x * blockDim.x + threadIdx.x;
    int l  = idx % LTOT;
    int ch = idx / LTOT;
    int s = l % 24, w = (l / 24) % 24, h = l / (24 * 24);

    float acc = bdw[ch];
    const float* wt  = wdw + ch * 27;
    const float* inc = in + (size_t)ch * LTOT;
    #pragma unroll
    for (int dh = -1; dh <= 1; ++dh) {
        int h2 = h + dh; if (h2 < 0 || h2 >= 24) continue;
        #pragma unroll
        for (int dw = -1; dw <= 1; ++dw) {
            int w2 = w + dw; if (w2 < 0 || w2 >= 24) continue;
            #pragma unroll
            for (int ds = -1; ds <= 1; ++ds) {
                int s2 = s + ds; if (s2 < 0 || s2 >= 24) continue;
                acc += wt[((dh + 1) * 3 + (dw + 1)) * 3 + (ds + 1)]
                     * inc[(h2 * 24 + w2) * 24 + s2];
            }
        }
    }
    out[idx] = acc;
}

// ---------------------------------------------------------------------------
// Local attention: one wave32 per (location l, head). Lanes 0..26 each own
// one of the 27 edge-clamped neighbors; softmax via wave32 shuffle reductions;
// output channels distributed across lanes (lane and lane+32 for lane<16).
// ---------------------------------------------------------------------------
__global__ __launch_bounds__(256) void attn_kernel(
    const float* __restrict__ qkv,  // [576][LTOT]  (q:0..191, k:192..383, v:384..575)
    float* __restrict__ out)        // [192][LTOT]
{
    const int lane = threadIdx.x & 31;
    const int wid  = blockIdx.x * 8 + (threadIdx.x >> 5);
    const int l    = wid >> 2;
    const int head = wid & 3;

    int s = l % 24, w = (l / 24) % 24, h = l / (24 * 24);
    int k  = (lane < 27) ? lane : 26;
    int h2 = h + (k / 9) - 1;       h2 = h2 < 0 ? 0 : (h2 > 23 ? 23 : h2);
    int w2 = w + ((k / 3) % 3) - 1; w2 = w2 < 0 ? 0 : (w2 > 23 ? 23 : w2);
    int s2 = s + (k % 3) - 1;       s2 = s2 < 0 ? 0 : (s2 > 23 ? 23 : s2);
    int lk = (h2 * 24 + w2) * 24 + s2;

    const float* q  = qkv + (size_t)(head * 48) * LTOT + l;
    const float* Kp = qkv + (size_t)(192 + head * 48) * LTOT + lk;
    float score = 0.f;
    #pragma unroll 8
    for (int j = 0; j < 48; ++j)
        score += q[(size_t)j * LTOT] * Kp[(size_t)j * LTOT];
    score *= 0.14433756729740643f;   // 48^-0.5
    if (lane >= 27) score = -1e30f;

    float m = score;
    #pragma unroll
    for (int off = 16; off; off >>= 1) m = fmaxf(m, __shfl_xor(m, off, 32));
    float p = (lane < 27) ? __expf(score - m) : 0.f;
    float ssum = p;
    #pragma unroll
    for (int off = 16; off; off >>= 1) ssum += __shfl_xor(ssum, off, 32);
    p /= ssum;

    float acc0 = 0.f, acc1 = 0.f;
    const float* Vb = qkv + (size_t)(384 + head * 48) * LTOT;
    for (int kk = 0; kk < 27; ++kk) {
        float pk  = __shfl(p, kk, 32);
        int   lkk = __shfl(lk, kk, 32);
        acc0 += pk * Vb[(size_t)lane * LTOT + lkk];
        if (lane < 16) acc1 += pk * Vb[(size_t)(lane + 32) * LTOT + lkk];
    }
    out[(size_t)(head * 48 + lane) * LTOT + l] = acc0;
    if (lane < 16) out[(size_t)(head * 48 + lane + 32) * LTOT + l] = acc1;
}

// ---------------------------------------------------------------------------
extern "C" void kernel_launch(void* const* d_in, const int* in_sizes, int n_in,
                              void* d_out, int out_size, void* d_ws, size_t ws_size,
                              hipStream_t stream) {
    const float* x      = (const float*)d_in[0];  // [192][13824]
    const float* w_qkv1 = (const float*)d_in[1];  // [576][192]
    const float* b_qkv1 = (const float*)d_in[2];  // [576]
    const float* w_dw   = (const float*)d_in[3];  // [576][27]
    const float* b_dw   = (const float*)d_in[4];  // [576]
    const float* w_proj = (const float*)d_in[5];  // [192][192]
    const float* b_proj = (const float*)d_in[6];  // [192]
    float* out = (float*)d_out;                   // [192][13824]

    float* qkv   = (float*)d_ws;                   // [576][13824]
    float* qkv2  = qkv  + (size_t)ODIM * LTOT;     // [576][13824]
    float* attnb = qkv2 + (size_t)ODIM * LTOT;     // [192][13824]

    // 1) QKV projection: 576x192 @ 192x13824
    wmma_gemm_kernel<<<dim3(LTOT / 64, ODIM / 64), 128, 0, stream>>>(
        x, w_qkv1, b_qkv1, qkv);

    // 2) Depthwise 3x3x3 conv (zero pad) over 576 channels
    dwconv_kernel<<<(ODIM * LTOT) / 256, 256, 0, stream>>>(qkv, w_dw, b_dw, qkv2);

    // 3) Local 27-neighbor attention (edge-clamped), 4 heads, d=48
    attn_kernel<<<(LTOT * 4) / 8, 256, 0, stream>>>(qkv2, attnb);

    // 4) Output projection: 192x192 @ 192x13824
    wmma_gemm_kernel<<<dim3(LTOT / 64, CDIM / 64), 128, 0, stream>>>(
        attnb, w_proj, b_proj, out);
}
